// AttLayer_16758962389013
// MI455X (gfx1250) — matrix operand order repaired and checked by reading, not compiled, not run on hardware
//
#include <hip/hip_runtime.h>
#include <hip/hip_bf16.h>

// ---------------- types ----------------
typedef __attribute__((ext_vector_type(16))) __bf16          bf16x16;
typedef __attribute__((ext_vector_type(8)))  float           f32x8;
typedef __attribute__((ext_vector_type(8)))  unsigned short  u16x8;

#define SCALE 0.03952847075210474f  // 640^-0.5

// fp32 -> bf16 (round to nearest even), storage as u16
__device__ __forceinline__ unsigned short f2bf(float f) {
    unsigned u = __float_as_uint(f);
    unsigned r = u + 0x7FFFu + ((u >> 16) & 1u);
    return (unsigned short)(r >> 16);
}

// A fragment 16x32 bf16: p = &A[(m0+row)*lda + hf*8] (row-major A)
__device__ __forceinline__ bf16x16 load_frag_a(const unsigned short* p) {
    union { bf16x16 bf; u16x8 h[2]; } u;
    u.h[0] = *(const u16x8*)(p);        // K = hf*8 + 0..7
    u.h[1] = *(const u16x8*)(p + 16);   // K = 16 + hf*8 + 0..7
    return u.bf;
}
// B fragment 32x16 bf16 from k-contiguous storage: p = &Bt[(n0+n)*ldb + hf*8]
__device__ __forceinline__ bf16x16 load_frag_b(const unsigned short* p) {
    union { bf16x16 bf; u16x8 h[2]; } u;
    u.h[0] = *(const u16x8*)(p);        // K = hf*8 + 0..7
    u.h[1] = *(const u16x8*)(p + 16);   // K = 16 + hf*8 + 0..7
    return u.bf;
}

#define WMMA_BF16(a, b, c) \
    __builtin_amdgcn_wmma_f32_16x16x32_bf16(false, (a), false, (b), (short)0, (c), false, false)

// ---- software-pipelined 32(M) x 64(N) x KK wave GEMM ----
// pa = &A[(m0+rw)*LDA + hf*8]   (A row-major, rows m0..m0+31)
// pb = &Bt[(n0+rw)*LDB + hf*8] (B k-contiguous per column, cols n0..n0+63)
template <int KK, int LDA, int LDB>
__device__ __forceinline__ void gemm_32x64(const unsigned short* __restrict__ pa,
                                           const unsigned short* __restrict__ pb,
                                           f32x8 (&acc)[2][4]) {
    bf16x16 a0 = load_frag_a(pa);
    bf16x16 a1 = load_frag_a(pa + 16 * LDA);
    bf16x16 b0 = load_frag_b(pb);
    bf16x16 b1 = load_frag_b(pb + 16 * LDB);
    bf16x16 b2 = load_frag_b(pb + 32 * LDB);
    bf16x16 b3 = load_frag_b(pb + 48 * LDB);
#pragma unroll
    for (int k = 32; k <= KK; k += 32) {
        bf16x16 na0, na1, nb0, nb1, nb2, nb3;
        if (k < KK) {                      // prefetch next iteration's fragments
            na0 = load_frag_a(pa + k);
            na1 = load_frag_a(pa + 16 * LDA + k);
            nb0 = load_frag_b(pb + k);
            nb1 = load_frag_b(pb + 16 * LDB + k);
            nb2 = load_frag_b(pb + 32 * LDB + k);
            nb3 = load_frag_b(pb + 48 * LDB + k);
        }
        acc[0][0] = WMMA_BF16(a0, b0, acc[0][0]);
        acc[1][0] = WMMA_BF16(a1, b0, acc[1][0]);
        acc[0][1] = WMMA_BF16(a0, b1, acc[0][1]);
        acc[1][1] = WMMA_BF16(a1, b1, acc[1][1]);
        acc[0][2] = WMMA_BF16(a0, b2, acc[0][2]);
        acc[1][2] = WMMA_BF16(a1, b2, acc[1][2]);
        acc[0][3] = WMMA_BF16(a0, b3, acc[0][3]);
        acc[1][3] = WMMA_BF16(a1, b3, acc[1][3]);
        a0 = na0; a1 = na1; b0 = nb0; b1 = nb1; b2 = nb2; b3 = nb3;
    }
}

// store one 16x16 f32 D-tile as bf16, row-major; dst = &out[m0*ld + n0]
__device__ __forceinline__ void store_bf16_tile(unsigned short* dst, int ld,
                                                const f32x8& c, int rw, int hf) {
    for (int r = 0; r < 8; ++r) dst[(size_t)(hf * 8 + r) * ld + rw] = f2bf(c[r]);
}

// ---------------- converts ----------------
__global__ void k_cvt_x(const float* __restrict__ x, unsigned short* __restrict__ xsb) {
    int idx = blockIdx.x * 256 + threadIdx.x;          // b*65536 + c*1024 + s
    int s = idx & 1023, c = (idx >> 10) & 63, b = idx >> 16;
    xsb[(size_t)b * 65536 + (size_t)s * 64 + c] = f2bf(x[idx]);
}
__global__ void k_cvt_wp(const float* __restrict__ W, unsigned short* __restrict__ WpT) {
    int idx = blockIdx.x * 256 + threadIdx.x;          // k*7680 + n
    int n = idx % 7680, k = idx / 7680;
    WpT[(size_t)n * 64 + k] = f2bf(W[idx]);
}
__global__ void k_cvt_wo(const float* __restrict__ W, unsigned short* __restrict__ WoT) {
    int idx = blockIdx.x * 256 + threadIdx.x;          // m*64 + c
    int c = idx & 63, m = idx >> 6;
    WoT[(size_t)c * 2560 + m] = f2bf(W[idx]);
}

// ---------------- GEMM1: proj = xs @ W_proj + b_proj ----------------
// grid (32, 120), block 32.  M=1024, N=7680, K=64.
__global__ __launch_bounds__(32) void k_proj(const unsigned short* __restrict__ xsb,
                                             const unsigned short* __restrict__ WpT,
                                             const float* __restrict__ b_proj,
                                             unsigned short* __restrict__ qkv,
                                             unsigned short* __restrict__ vT) {
    const int lane = threadIdx.x & 31, rw = lane & 15, hf = lane >> 4;
    const int m0 = blockIdx.x * 32, n0 = blockIdx.y * 64;
    f32x8 acc[2][4];
    for (int t = 0; t < 4; ++t) {
        float bv = b_proj[n0 + t * 16 + rw];
        for (int r = 0; r < 8; ++r) { acc[0][t][r] = bv; acc[1][t][r] = bv; }
    }
    gemm_32x64<64, 64, 64>(xsb + (size_t)(m0 + rw) * 64 + hf * 8,
                           WpT + (size_t)(n0 + rw) * 64 + hf * 8, acc);
    for (int t = 0; t < 4; ++t) {
        int col = n0 + t * 16 + rw;
        int m1920 = col % 1920;
        if (m1920 < 1280) {                               // q/k: row-major into qkv
            for (int mm = 0; mm < 2; ++mm)
                store_bf16_tile(qkv + (size_t)(m0 + mm * 16) * 7680 + n0 + t * 16,
                                7680, acc[mm][t], rw, hf);
        } else {                                          // v: transposed vT[h*640+d][s]
            int h = col / 1920, d = m1920 - 1280;
            for (int mm = 0; mm < 2; ++mm) {
                u16x8 o;
                for (int r = 0; r < 8; ++r) o[r] = f2bf(acc[mm][t][r]);
                *(u16x8*)(vT + (size_t)(h * 640 + d) * 1024 + m0 + mm * 16 + hf * 8) = o;
            }
        }
    }
}

// ---------------- GEMM2: S[h][i][j] = Q_h @ K_h^T ----------------
// grid (32, 16, 4), block 32.  K=640.
__global__ __launch_bounds__(32) void k_scores(const unsigned short* __restrict__ qkv,
                                               float* __restrict__ S) {
    const int lane = threadIdx.x & 31, rw = lane & 15, hf = lane >> 4;
    const int i0 = blockIdx.x * 32, j0 = blockIdx.y * 64, h = blockIdx.z;
    f32x8 acc[2][4] = {};
    gemm_32x64<640, 7680, 7680>(qkv + (size_t)(i0 + rw) * 7680 + h * 1920 + hf * 8,
                                qkv + (size_t)(j0 + rw) * 7680 + h * 1920 + 640 + hf * 8, acc);
    for (int mm = 0; mm < 2; ++mm)
        for (int t = 0; t < 4; ++t) {
            float* dst = S + ((size_t)h << 20) + (size_t)(i0 + mm * 16 + hf * 8) * 1024
                           + j0 + t * 16 + rw;
            for (int r = 0; r < 8; ++r) dst[(size_t)r * 1024] = acc[mm][t][r];
        }
}

// ---------------- softmax over i (columns of S[i][j]) ----------------
// pass A: per (chunk,h,j): local max + sum of exp.  grid (16, 8), block 256.
__global__ void k_soft_part(const float* __restrict__ S,
                            float* __restrict__ pm, float* __restrict__ ps) {
    int cidx = blockIdx.x * 256 + threadIdx.x;     // h*1024 + j
    int h = cidx >> 10, j = cidx & 1023;
    const float* col = S + ((size_t)h << 20) + (size_t)blockIdx.y * 128 * 1024 + j;
    float m = -INFINITY;
    for (int i = 0; i < 128; ++i) m = fmaxf(m, col[(size_t)i * 1024]);
    float s = 0.f;
    for (int i = 0; i < 128; ++i) s += __expf(SCALE * (col[(size_t)i * 1024] - m));
    pm[blockIdx.y * 4096 + cidx] = m;
    ps[blockIdx.y * 4096 + cidx] = s;
}
// pass B: combine 8 chunks.  grid 16, block 256.
__global__ void k_soft_comb(const float* __restrict__ pm, const float* __restrict__ ps,
                            float* __restrict__ cmax, float* __restrict__ crcp) {
    int idx = blockIdx.x * 256 + threadIdx.x;      // 0..4095
    float M = -INFINITY;
    for (int c = 0; c < 8; ++c) M = fmaxf(M, pm[c * 4096 + idx]);
    float s = 0.f;
    for (int c = 0; c < 8; ++c) s += ps[c * 4096 + idx] * __expf(SCALE * (pm[c * 4096 + idx] - M));
    cmax[idx] = M;
    crcp[idx] = 1.0f / s;
}
// pass C: P = bf16(exp(scale*(S-m))/sum).  grid 16384, block 256.
__global__ void k_soft_norm(const float* __restrict__ S, const float* __restrict__ cmax,
                            const float* __restrict__ crcp, unsigned short* __restrict__ P) {
    int idx = blockIdx.x * 256 + threadIdx.x;      // h<<20 | i<<10 | j
    int h = idx >> 20, j = idx & 1023;
    int cj = (h << 10) | j;
    P[idx] = f2bf(__expf(SCALE * (S[idx] - cmax[cj])) * crcp[cj]);
}

// ---------------- GEMM3: att[i][h*640+d] = P_h @ V_h ----------------
// grid (32, 10, 4), block 32.  K=1024.
__global__ __launch_bounds__(32) void k_att(const unsigned short* __restrict__ P,
                                            const unsigned short* __restrict__ vT,
                                            unsigned short* __restrict__ att) {
    const int lane = threadIdx.x & 31, rw = lane & 15, hf = lane >> 4;
    const int i0 = blockIdx.x * 32, d0 = blockIdx.y * 64, h = blockIdx.z;
    f32x8 acc[2][4] = {};
    gemm_32x64<1024, 1024, 1024>(P + ((size_t)h << 20) + (size_t)(i0 + rw) * 1024 + hf * 8,
                                 vT + (size_t)(h * 640 + d0 + rw) * 1024 + hf * 8, acc);
    for (int mm = 0; mm < 2; ++mm)
        for (int t = 0; t < 4; ++t)
            store_bf16_tile(att + (size_t)(i0 + mm * 16) * 2560 + h * 640 + d0 + t * 16,
                            2560, acc[mm][t], rw, hf);
}

// ---------------- GEMM4: out[b][c][s] = att @ W_out + b_out + x[b][c][s] ----------------
// grid (32, 4), block 32.  TM=32, TN=16, K=2560.
__global__ __launch_bounds__(32) void k_out(const unsigned short* __restrict__ att,
                                            const unsigned short* __restrict__ WoT,
                                            const float* __restrict__ b_out,
                                            const float* __restrict__ xb,
                                            float* __restrict__ outb) {
    const int lane = threadIdx.x & 31, rw = lane & 15, hf = lane >> 4;
    const int i0 = blockIdx.x * 32, c0 = blockIdx.y * 16;
    float bv = b_out[c0 + rw];
    f32x8 acc0, acc1;
    for (int r = 0; r < 8; ++r) { acc0[r] = bv; acc1[r] = bv; }
    const unsigned short* pa = att + (size_t)(i0 + rw) * 2560 + hf * 8;
    const unsigned short* pb = WoT + (size_t)(c0 + rw) * 2560 + hf * 8;
#pragma unroll 4
    for (int k = 0; k < 2560; k += 32) {
        bf16x16 a0 = load_frag_a(pa + k);
        bf16x16 a1 = load_frag_a(pa + 16 * 2560 + k);
        bf16x16 b  = load_frag_b(pb + k);
        acc0 = WMMA_BF16(a0, b, acc0);
        acc1 = WMMA_BF16(a1, b, acc1);
    }
    int c = c0 + rw;
    {   // rows i0 .. i0+15
        int s0 = i0 + hf * 8;
        f32x8 o = acc0 + *(const f32x8*)(xb + (size_t)c * 1024 + s0);
        *(f32x8*)(outb + (size_t)c * 1024 + s0) = o;
    }
    {   // rows i0+16 .. i0+31
        int s0 = i0 + 16 + hf * 8;
        f32x8 o = acc1 + *(const f32x8*)(xb + (size_t)c * 1024 + s0);
        *(f32x8*)(outb + (size_t)c * 1024 + s0) = o;
    }
}

// ---------------- host ----------------
extern "C" void kernel_launch(void* const* d_in, const int* in_sizes, int n_in,
                              void* d_out, int out_size, void* d_ws, size_t ws_size,
                              hipStream_t stream) {
    (void)in_sizes; (void)n_in; (void)out_size;
    const float* x      = (const float*)d_in[0];
    const float* W_proj = (const float*)d_in[2];
    const float* b_proj = (const float*)d_in[3];
    const float* W_out  = (const float*)d_in[4];
    const float* b_out  = (const float*)d_in[5];
    float* out = (float*)d_out;

    char* ws = (char*)d_ws;
    size_t off = 0;
    auto alloc = [&](size_t bytes) -> char* {
        char* p = ws + off;
        off = (off + bytes + 255) & ~(size_t)255;
        return p;
    };
    unsigned short* xsb = (unsigned short*)alloc((size_t)8 * 1024 * 64 * 2);
    unsigned short* WpT = (unsigned short*)alloc((size_t)7680 * 64 * 2);
    unsigned short* WoT = (unsigned short*)alloc((size_t)64 * 2560 * 2);
    unsigned short* qkv = (unsigned short*)alloc((size_t)1024 * 7680 * 2);   // per-batch reuse
    unsigned short* vT  = (unsigned short*)alloc((size_t)4 * 640 * 1024 * 2);
    float*          S   = (float*)alloc((size_t)4 * 1024 * 1024 * 4);
    unsigned short* P   = (unsigned short*)alloc((size_t)4 * 1024 * 1024 * 2);
    unsigned short* att = (unsigned short*)alloc((size_t)1024 * 2560 * 2);
    float*          pm  = (float*)alloc((size_t)8 * 4096 * 4);
    float*          ps  = (float*)alloc((size_t)8 * 4096 * 4);
    float*          cm  = (float*)alloc(4096 * 4);
    float*          cr  = (float*)alloc(4096 * 4);
    if (off > ws_size) return;   // ~54 MB needed

    k_cvt_x <<<2048, 256, 0, stream>>>(x, xsb);
    k_cvt_wp<<<1920, 256, 0, stream>>>(W_proj, WpT);
    k_cvt_wo<<< 640, 256, 0, stream>>>(W_out, WoT);

    for (int b = 0; b < 8; ++b) {
        const unsigned short* xsb_b = xsb + (size_t)b * 1024 * 64;
        k_proj     <<<dim3(32, 120),   32, 0, stream>>>(xsb_b, WpT, b_proj, qkv, vT);
        k_scores   <<<dim3(32, 16, 4), 32, 0, stream>>>(qkv, S);
        k_soft_part<<<dim3(16, 8),    256, 0, stream>>>(S, pm, ps);
        k_soft_comb<<<16,             256, 0, stream>>>(pm, ps, cm, cr);
        k_soft_norm<<<16384,          256, 0, stream>>>(S, cm, cr, P);
        k_att      <<<dim3(32, 10, 4), 32, 0, stream>>>(P, vT, att);
        k_out      <<<dim3(32, 4),     32, 0, stream>>>(att, WoT, b_out,
                                                        x + (size_t)b * 65536,
                                                        out + (size_t)b * 65536);
    }
}
